// SwitchFeedForward_41918880809529
// MI455X (gfx1250) — compile-verified, matrix operand order checked by reading, not run on hardware
//
#include <hip/hip_runtime.h>
#include <hip/hip_bf16.h>
#include <math.h>

#define D_MODEL 1024
#define N_FF    4096
#define N_EXP   8
#define NTOK    8192
#define CAP     1280          // int(8192 * 1.25 / 8)
#define LDSB    40            // padded LDS row stride (bf16 elems) = 80B, 16B-aligned

#if __has_builtin(__builtin_amdgcn_tensor_load_to_lds)
#define USE_TDM 1
#else
#define USE_TDM 0
#endif

typedef __attribute__((ext_vector_type(16))) __bf16 v16bf;
typedef __attribute__((ext_vector_type(8)))  __bf16 v8bf;
typedef __attribute__((ext_vector_type(8)))  float  v8f;
typedef __attribute__((ext_vector_type(4)))  unsigned int v4u;
typedef __attribute__((ext_vector_type(8)))  int v8i;
typedef __attribute__((ext_vector_type(4)))  int v4i;

__device__ __forceinline__ unsigned pack2bf(float a, float b) {
    union { __bf16 h[2]; unsigned u; } z;
    z.h[0] = (__bf16)a; z.h[1] = (__bf16)b;
    return z.u;
}

// -------- router: logits -> softmax -> argmax; per-block hist + prob sums ---
__global__ __launch_bounds__(256) void k_router(const float* __restrict__ x,
        const float* __restrict__ Wsw, const float* __restrict__ bsw,
        int* __restrict__ route, int* __restrict__ bcnt, float* __restrict__ bprob)
{
    __shared__ float sW[D_MODEL * N_EXP];   // 32KB
    __shared__ float sP[256 * N_EXP];       // 8KB
    __shared__ int   sCnt[N_EXP];
    const int tid = threadIdx.x;
    for (int i = tid; i < D_MODEL * N_EXP; i += 256) sW[i] = Wsw[i];
    if (tid < N_EXP) sCnt[tid] = 0;
    __syncthreads();

    const int n = blockIdx.x * 256 + tid;
    const float* xr = x + (size_t)n * D_MODEL;
    float acc[N_EXP];
    #pragma unroll
    for (int e = 0; e < N_EXP; ++e) acc[e] = bsw[e];
    for (int d = 0; d < D_MODEL; ++d) {
        float xv = xr[d];
        #pragma unroll
        for (int e = 0; e < N_EXP; ++e) acc[e] = fmaf(xv, sW[d * N_EXP + e], acc[e]);
    }
    float m = acc[0]; int r = 0;
    #pragma unroll
    for (int e = 1; e < N_EXP; ++e) if (acc[e] > m) { m = acc[e]; r = e; }  // first-max tie break
    float s = 0.f, p[N_EXP];
    #pragma unroll
    for (int e = 0; e < N_EXP; ++e) { p[e] = __expf(acc[e] - m); s += p[e]; }
    float is = 1.f / s;
    #pragma unroll
    for (int e = 0; e < N_EXP; ++e) sP[tid * N_EXP + e] = p[e] * is;
    route[n] = r;
    atomicAdd(&sCnt[r], 1);
    __syncthreads();
    if (tid < N_EXP) {
        bcnt[blockIdx.x * N_EXP + tid] = sCnt[tid];
        float t = 0.f;
        for (int i = 0; i < 256; ++i) t += sP[i * N_EXP + tid];  // deterministic order
        bprob[blockIdx.x * N_EXP + tid] = t;
    }
}

// -------- scan: exclusive block offsets, totals, counts/probsum/ndrop out ---
__global__ void k_scan(const int* __restrict__ bcnt, const float* __restrict__ bprob,
                       int* __restrict__ boff, int* __restrict__ cnts,
                       float* __restrict__ out_counts, float* __restrict__ out_ps,
                       int* __restrict__ out_nd)
{
    const int tid = threadIdx.x;
    if (tid < N_EXP) {
        int off = 0;
        for (int b = 0; b < 32; ++b) { boff[b * N_EXP + tid] = off; off += bcnt[b * N_EXP + tid]; }
        cnts[tid] = off;
        out_counts[tid] = (float)off;
        float s = 0.f;
        for (int b = 0; b < 32; ++b) s += bprob[b * N_EXP + tid];
        out_ps[tid] = s;
    }
    __syncthreads();
    if (tid == 0) {
        int kept = 0;
        for (int e = 0; e < N_EXP; ++e) { int c = cnts[e]; kept += (c < CAP ? c : CAP); }
        *out_nd = NTOK - kept;
    }
}

// -------- rank: ordered position within expert queue (token-index order) ----
__global__ __launch_bounds__(256) void k_rank(const int* __restrict__ route,
        const int* __restrict__ boff, int* __restrict__ dslot, int* __restrict__ inv)
{
    __shared__ int woff[8][N_EXP];
    const int tid = threadIdx.x, wave = tid >> 5, lane = tid & 31;
    const int n = blockIdx.x * 256 + tid;
    const int r = route[n];
    unsigned bm8[N_EXP];
    #pragma unroll
    for (int e = 0; e < N_EXP; ++e) bm8[e] = (unsigned)__ballot(r == e);
    if (lane < N_EXP) woff[wave][lane] = __popc(bm8[lane]);
    const int rIn = __popc(bm8[r] & ((1u << lane) - 1u));
    __syncthreads();
    if (tid < N_EXP) {
        int o = 0;
        for (int w = 0; w < 8; ++w) { int t = woff[w][tid]; woff[w][tid] = o; o += t; }
    }
    __syncthreads();
    const int pos = boff[blockIdx.x * N_EXP + r] + woff[wave][r] + rIn;
    const int slot = (pos < CAP) ? (r * CAP + pos) : -1;
    dslot[n] = slot;
    if (slot >= 0) inv[slot] = n;
}

// -------- dispatch: wave per token, fp32 -> packed bf16 into disp ----------
__global__ __launch_bounds__(256) void k_dispatch(const float* __restrict__ x,
        const int* __restrict__ dslot, __bf16* __restrict__ disp)
{
    const int tok  = (blockIdx.x << 3) + (threadIdx.x >> 5);
    const int lane = threadIdx.x & 31;
    const int slot = dslot[tok];
    if (slot < 0) return;
    const float2* src = (const float2*)(x + (size_t)tok * D_MODEL);
    unsigned* dst = (unsigned*)(disp + (size_t)slot * D_MODEL);
    #pragma unroll
    for (int i = 0; i < D_MODEL / 64; ++i) {   // 16 iters
        float2 v = src[lane + i * 32];
        dst[lane + i * 32] = pack2bf(v.x, v.y);
    }
}

// -------- prefill out = x (dropped tokens pass through) --------------------
__global__ __launch_bounds__(256) void k_prefill(const float4* __restrict__ x,
                                                 float4* __restrict__ out)
{
    const int total = NTOK * D_MODEL / 4;
    for (int i = blockIdx.x * 256 + threadIdx.x; i < total; i += 2048 * 256)
        out[i] = x[i];
}

// -------- WMMA bf16 fragment load per ISA 16-bit A/B layout ----------------
__device__ __forceinline__ v16bf load_frag(const __bf16* base, int row0, int lane) {
    const __bf16* p = base + (row0 + (lane & 15)) * LDSB + ((lane >> 4) << 3);
    v8bf a = *(const v8bf*)p;          // K = off .. off+7
    v8bf b = *(const v8bf*)(p + 16);   // K = off+16 .. off+23
    return __builtin_shufflevector(a, b, 0,1,2,3,4,5,6,7,8,9,10,11,12,13,14,15);
}

// -------- grouped GEMM: 128x128 tile, 8 waves, 4x2 WMMA accums per wave ----
// Double-buffered pipeline: TDM streams next A-tile + threads stage next
// W-tile while WMMAs run on the current buffers; one barrier per K-step.
// FUSE=false: Hout = gelu(A@W + bias) as bf16 [E, CAP, N]
// FUSE=true : Yout[tok] = A@W + bias   scattered to tokens via inv[E*CAP]
template<bool FUSE>
__global__ __launch_bounds__(256) void k_gemm(
    const __bf16* __restrict__ A, const float* __restrict__ W,
    const float* __restrict__ bias, const int* __restrict__ cnts,
    __bf16* __restrict__ Hout, float* __restrict__ Yout,
    const int* __restrict__ inv, int K, int N)
{
    const int e  = blockIdx.z;
    const int m0 = blockIdx.y * 128;
    const int n0 = blockIdx.x * 128;
    const int ce = cnts[e];
    const int mlim = (ce < CAP ? ce : CAP);
    if (m0 >= mlim) return;                    // skip tiles past expert load

    __shared__ __bf16 As[2][128 * LDSB];       // 2 x 10KB (TDM fills with pad)
    __shared__ __bf16 Bs[2][128 * LDSB];       // 2 x 10KB

    const int tid  = threadIdx.x;
    const int lane = tid & 31;
    const int wave = tid >> 5;
    const int mb = (wave & 1) * 64;            // wave sub-tile: 64(M) x 32(N)
    const int nb = (wave >> 1) * 32;

    v8f acc[4][2];
    #pragma unroll
    for (int i = 0; i < 4; ++i)
        #pragma unroll
        for (int j = 0; j < 2; ++j)
            #pragma unroll
            for (int r = 0; r < 8; ++r) acc[i][j][r] = 0.f;

    const __bf16* Ag = A + ((size_t)e * CAP + m0) * (size_t)K;
    const float*  Wg = W + (size_t)e * K * (size_t)N + n0;

    // B-tile roles: thread handles k-pair kp (k=2kp,2kp+1) x 8 cols at ng*8
    const int kp = tid & 15;
    const int ng = tid >> 4;

    float4 w00, w01, w10, w11;                 // staged W tile (registers)
    auto loadW = [&](int kb) {
        const float4* p0 = (const float4*)(Wg + (size_t)(kb + 2 * kp) * N + ng * 8);
        const float4* p1 = (const float4*)(Wg + (size_t)(kb + 2 * kp + 1) * N + ng * 8);
        w00 = p0[0]; w01 = p0[1]; w10 = p1[0]; w11 = p1[1];
    };
    auto storeW = [&](int buf) {
        float r0[8] = { w00.x, w00.y, w00.z, w00.w, w01.x, w01.y, w01.z, w01.w };
        float r1[8] = { w10.x, w10.y, w10.z, w10.w, w11.x, w11.y, w11.z, w11.w };
        #pragma unroll
        for (int i = 0; i < 8; ++i)
            *(unsigned*)&Bs[buf][(ng * 8 + i) * LDSB + 2 * kp] = pack2bf(r0[i], r1[i]);
    };

#if USE_TDM
    // ---- Tensor DMA descriptor (2D tile 32(k) x 128(m), bf16, padded LDS) ----
    // group1: data_size=1(2B) | pad_enable | pad_interval=3(16 DW=64B row)
    //         | pad_amount=3(4 DW=16B -> row stride 40 elems == LDSB)
    const unsigned ldsA0 = (unsigned)(size_t)&As[0][0];
    const unsigned ldsA1 = (unsigned)(size_t)&As[1][0];
    v8i g1;
    g1[0] = 0x06D10000;                          // mask=0, ds=2B, pad cfg
    g1[1] = (int)((unsigned)K << 16);            // abar=0 | tensor_dim0 lo16
    g1[2] = (int)(((unsigned)K >> 16) & 0xFFFFu) | (128 << 16); // dim0 hi | dim1 lo
    g1[3] = (32 << 16);                          // dim1 hi | tile_dim0=32
    g1[4] = 128;                                 // tile_dim1=128, tile_dim2=0
    g1[5] = K;                                   // tensor_dim0_stride lo32
    g1[6] = 0;
    g1[7] = 0;
    const v4i gz4 = {0, 0, 0, 0};
    const v8i gz8 = {0, 0, 0, 0, 0, 0, 0, 0};
    const unsigned long long gaBase = (unsigned long long)(size_t)Ag;
    auto tdmIssue = [&](int kb, int buf) {
        const unsigned long long ga = gaBase + (unsigned)(kb * 2);
        v4u g0;
        g0[0] = 1u;                                           // count=1
        g0[1] = buf ? ldsA1 : ldsA0;                          // lds_addr
        g0[2] = (unsigned)ga;                                 // global lo
        g0[3] = ((unsigned)(ga >> 32) & 0x01FFFFFFu) | 0x80000000u; // hi|type=2
        __builtin_amdgcn_tensor_load_to_lds(g0, g1, gz4, gz4, gz8, 0);
    };
#else
    uint4 av0, av1;                             // staged A tile (registers)
    const int ar = tid >> 1, ah = (tid & 1) * 16;
    auto loadA = [&](int kb) {
        const uint4* ap = (const uint4*)(Ag + (size_t)ar * K + kb + ah);
        av0 = ap[0]; av1 = ap[1];
    };
    auto storeA = [&](int buf) {
        *(uint4*)(As[buf] + ar * LDSB + ah)     = av0;
        *(uint4*)(As[buf] + ar * LDSB + ah + 8) = av1;
    };
#endif

    // ---- prologue: stage tile 0 into buffer 0 ----
#if USE_TDM
    if (wave == 0) tdmIssue(0, 0);
#else
    loadA(0);
#endif
    loadW(0);
    storeW(0);
#if !USE_TDM
    storeA(0);
#endif
#if USE_TDM
    if (wave == 0) __builtin_amdgcn_s_wait_tensorcnt(0);
#endif
    __syncthreads();

    for (int kb = 0; kb < K; kb += 32) {
        const int P = (kb >> 5) & 1;
        const bool hasNext = (kb + 32) < K;
        if (hasNext) {
            // start next tile's traffic before compute (hidden under WMMAs)
#if USE_TDM
            if (wave == 0) tdmIssue(kb + 32, 1 - P);
#else
            loadA(kb + 32);
#endif
            loadW(kb + 32);
            if (kb + 64 < K)
                __builtin_prefetch((const void*)(Wg + (size_t)(kb + 64 + 2 * kp) * N + ng * 8), 0, 1);
        }

        // compute on buffer P
        v16bf af[4], bfv[2];
        #pragma unroll
        for (int i = 0; i < 4; ++i) af[i]  = load_frag(As[P], mb + i * 16, lane);
        #pragma unroll
        for (int j = 0; j < 2; ++j) bfv[j] = load_frag(Bs[P], nb + j * 16, lane);
        #pragma unroll
        for (int i = 0; i < 4; ++i)
            #pragma unroll
            for (int j = 0; j < 2; ++j)
                acc[i][j] = __builtin_amdgcn_wmma_f32_16x16x32_bf16(
                    false, af[i], false, bfv[j], (short)0, acc[i][j], false, false);

        if (hasNext) {
            storeW(1 - P);                      // convert + publish next W tile
#if !USE_TDM
            storeA(1 - P);
#endif
#if USE_TDM
            if (wave == 0) __builtin_amdgcn_s_wait_tensorcnt(0);  // next A landed
#endif
            __syncthreads();                    // one barrier per K-step
        }
    }

    // epilogue: C/D layout -> lanes 0-15 rows r, lanes 16-31 rows r+8
    const int colLane = lane & 15;
    const int rowHalf = (lane >> 4) * 8;
    float bv[2];
    #pragma unroll
    for (int j = 0; j < 2; ++j) bv[j] = bias[e * N + n0 + nb + j * 16 + colLane];

    #pragma unroll
    for (int i = 0; i < 4; ++i) {
        #pragma unroll
        for (int r = 0; r < 8; ++r) {
            const int row = m0 + mb + i * 16 + rowHalf + r;
            int tok = 0;
            if constexpr (FUSE) { tok = inv[e * CAP + row]; if (tok < 0) continue; }
            #pragma unroll
            for (int j = 0; j < 2; ++j) {
                const int col = n0 + nb + j * 16 + colLane;
                float v = acc[i][j][r] + bv[j];
                if constexpr (!FUSE) {
                    float g = 0.5f * v * (1.0f + erff(v * 0.70710678118f));  // exact gelu
                    Hout[((size_t)e * CAP + row) * (size_t)N + col] = (__bf16)g;
                } else {
                    Yout[(size_t)tok * D_MODEL + col] = v;
                }
            }
        }
    }
}

extern "C" void kernel_launch(void* const* d_in, const int* in_sizes, int n_in,
                              void* d_out, int out_size, void* d_ws, size_t ws_size,
                              hipStream_t stream)
{
    const float* x   = (const float*)d_in[0];
    const float* Wsw = (const float*)d_in[1];
    const float* bsw = (const float*)d_in[2];
    const float* W1  = (const float*)d_in[3];
    const float* b1  = (const float*)d_in[4];
    const float* W2  = (const float*)d_in[5];
    const float* b2  = (const float*)d_in[6];

    char* ws = (char*)d_ws;
    int*    route = (int*)(ws + 0);                 // 32768 B
    int*    dslot = (int*)(ws + 32768);             // 32768 B
    int*    bcnt  = (int*)(ws + 65536);             // 1024 B
    float*  bprob = (float*)(ws + 66560);           // 1024 B
    int*    boff  = (int*)(ws + 67584);             // 1024 B
    int*    cnts  = (int*)(ws + 68608);             // 32 B
    int*    inv   = (int*)(ws + 69632);             // 8*1280*4 = 40960 B
    __bf16* disp  = (__bf16*)(ws + 131072);         // 8*1280*1024*2 = 20,971,520 B
    __bf16* H     = (__bf16*)(ws + 131072 + 20971520); // 8*1280*4096*2 = 83,886,080 B

    float* out     = (float*)d_out;
    float* out_cnt = out + (size_t)NTOK * D_MODEL;
    float* out_ps  = out_cnt + N_EXP;
    int*   out_nd  = (int*)(out_ps + N_EXP);

    (void)hipMemsetAsync(disp, 0, (size_t)N_EXP * CAP * D_MODEL * sizeof(__bf16), stream);
    (void)hipMemsetAsync(inv, 0xFF, (size_t)N_EXP * CAP * sizeof(int), stream);

    k_router  <<<32,   256, 0, stream>>>(x, Wsw, bsw, route, bcnt, bprob);
    k_scan    <<<1,    64,  0, stream>>>(bcnt, bprob, boff, cnts, out_cnt, out_ps, out_nd);
    k_rank    <<<32,   256, 0, stream>>>(route, boff, dslot, inv);
    k_dispatch<<<NTOK / 8, 256, 0, stream>>>(x, dslot, disp);
    k_prefill <<<2048, 256, 0, stream>>>((const float4*)x, (float4*)out);
    k_gemm<false><<<dim3(N_FF / 128,    CAP / 128, N_EXP), 256, 0, stream>>>(
        disp, W1, b1, cnts, H, nullptr, nullptr, D_MODEL, N_FF);
    k_gemm<true> <<<dim3(D_MODEL / 128, CAP / 128, N_EXP), 256, 0, stream>>>(
        H, W2, b2, cnts, nullptr, out, inv, N_FF, D_MODEL);
}
